// EncoderDecoder_46600395161679
// MI455X (gfx1250) — matrix operand order repaired and checked by reading, not compiled, not run on hardware
//
#include <hip/hip_runtime.h>
#include <hip/hip_bf16.h>
#include <math.h>

// ---------------- model dims (compile-time constants) ----------------
#define BB   32
#define SS   400
#define LL   50
#define EE   256
#define HH   512
#define AA   64
#define VV   50000

typedef __attribute__((ext_vector_type(16))) __bf16 bf16x16;
typedef __attribute__((ext_vector_type(8)))  float  f32x8;

static __device__ __forceinline__ __bf16 f2bf(float f) { return (__bf16)f; }
static __device__ __forceinline__ float sigm(float x) { return 1.0f / (1.0f + __expf(-x)); }

static __device__ __forceinline__ void load_a_frag(bf16x16& a, const float* Ar,
                                                   int k0, int hi)
{
#pragma unroll
    for (int v = 0; v < 8; ++v) {
        // 16-bit A-matrix 16x32 layout: VGPR v holds a K pair; lanes 16-31 offset +8
        int kk = k0 + ((v < 4) ? 2 * v : 8 + 2 * v) + (hi ? 8 : 0);
        float2 f = *(const float2*)(Ar + kk);
        a[2 * v]     = f2bf(f.x);
        a[2 * v + 1] = f2bf(f.y);
    }
}

static __device__ __forceinline__ void load_b_frag(bf16x16& b, const __bf16* Br,
                                                   int k0, int hi)
{
#pragma unroll
    for (int v = 0; v < 8; ++v) {
        // B 32x16: same K layout as A (VGPR v holds a K pair; lanes 16-31 offset +8)
        int kk = k0 + ((v < 4) ? 2 * v : 8 + 2 * v) + (hi ? 8 : 0);
        b[2 * v]     = Br[kk];
        b[2 * v + 1] = Br[kk + 1];
    }
}

// ---------------------------------------------------------------------
// Register-blocked WMMA GEMM:
//   C[M,N] = act( A[M,K](f32) @ Bw[N,K](bf16)^T + bias )
// One wave computes an (MT*16) x (NT*16) block: MT*NT accumulators,
// each A fragment reused NT times, each B fragment reused MT times.
// M must be a multiple of MT*16; K a multiple of 32; ragged N edge is
// handled with wave-uniform subtile guards (EXEC stays all-ones).
// revS/revB: optional row remap for the backward-encoder input GEMM.
// ---------------------------------------------------------------------
template<int MT, int NT>
__global__ void k_wmma_gemm(const float* __restrict__ A, int lda,
                            const __bf16* __restrict__ Bw,
                            const float* __restrict__ bias,
                            float* __restrict__ C, int ldc,
                            int M, int N, int K, int act, int revS, int revB)
{
    const int wave   = threadIdx.x >> 5;
    const int lane   = threadIdx.x & 31;
    const int lane15 = lane & 15;
    const int hi     = (lane >= 16) ? 1 : 0;

    const int blocksN = (N + NT * 16 - 1) / (NT * 16);
    const int blocksM = M / (MT * 16);
    const int wt = blockIdx.x * (blockDim.x >> 5) + wave;
    if (wt >= blocksM * blocksN) return;            // wave-uniform exit
    const int bm = wt / blocksN;
    const int bn = wt % blocksN;

    const float* Ar[MT];
#pragma unroll
    for (int mt = 0; mt < MT; ++mt) {
        int row = bm * MT * 16 + mt * 16 + lane15;
        if (revS) { int rs = row / revB, rb = row % revB; row = (revS - 1 - rs) * revB + rb; }
        Ar[mt] = A + (size_t)row * lda;
    }
    const __bf16* Br[NT];
    int ncol[NT];
#pragma unroll
    for (int nt = 0; nt < NT; ++nt) {
        ncol[nt] = bn * NT * 16 + nt * 16;
        Br[nt] = Bw + (size_t)(ncol[nt] + lane15) * K;
    }

    f32x8 acc[MT][NT];
    const f32x8 zero = {};
#pragma unroll
    for (int mt = 0; mt < MT; ++mt)
#pragma unroll
        for (int nt = 0; nt < NT; ++nt) acc[mt][nt] = zero;

    for (int k0 = 0; k0 < K; k0 += 32) {
        __builtin_prefetch(Br[0] + k0 + 128, 0, 1);   // global_prefetch of weight stream
        bf16x16 a[MT];
#pragma unroll
        for (int mt = 0; mt < MT; ++mt) load_a_frag(a[mt], Ar[mt], k0, hi);
        bf16x16 b[NT];
#pragma unroll
        for (int nt = 0; nt < NT; ++nt) {
            if (ncol[nt] < N) load_b_frag(b[nt], Br[nt], k0, hi);
            else {
#pragma unroll
                for (int q = 0; q < 16; ++q) b[nt][q] = (__bf16)0.0f;
            }
        }
#pragma unroll
        for (int mt = 0; mt < MT; ++mt)
#pragma unroll
            for (int nt = 0; nt < NT; ++nt)
                if (ncol[nt] < N)
                    acc[mt][nt] = __builtin_amdgcn_wmma_f32_16x16x32_bf16(
                        false, a[mt], false, b[nt], (short)0, acc[mt][nt], false, false);
    }

#pragma unroll
    for (int nt = 0; nt < NT; ++nt) {
        if (ncol[nt] >= N) continue;
        const int col = ncol[nt] + lane15;
        const float bv = bias ? bias[col] : 0.0f;
#pragma unroll
        for (int mt = 0; mt < MT; ++mt) {
#pragma unroll
            for (int v = 0; v < 8; ++v) {
                int r2 = bm * MT * 16 + mt * 16 + v + (hi ? 8 : 0);
                float x = acc[mt][nt][v] + bv;
                if (act == 1) x = tanhf(x);
                C[(size_t)r2 * ldc + col] = x;
            }
        }
    }
}

// ---------------------------------------------------------------------
// Fused bidirectional GRU encoder scan.
// grid = 2 blocks (dir 0 = fwd, dir 1 = bwd), 1024 threads = 32 waves.
// h (32x512 f32) and gh (32x1536 f32) live in LDS (256 KB dynamic).
// gi (x@Wih^T+bih) is precomputed by k_wmma_gemm in (s,b) row order.
// Recurrent gh = h@Whh^T via WMMA: each wave owns a 32x48 block
// (2 M-tiles x 3 N-tiles), sharing A fragments (LDS h) across 3 tiles
// and B fragments (Whh) across 2 tiles -> 6 WMMAs per 5 fragment loads.
// ---------------------------------------------------------------------
__global__ void k_enc_scan(const float* __restrict__ gi_f, const float* __restrict__ gi_b,
                           const __bf16* __restrict__ Whh_f, const __bf16* __restrict__ Whh_b,
                           const float* __restrict__ bhh_f, const float* __restrict__ bhh_b,
                           float* __restrict__ enc_hid, float* __restrict__ enc_final)
{
    extern __shared__ float smem[];
    float* h_s  = smem;                 // 32*512
    float* gh_s = smem + BB * HH;       // 32*1536

    const int dir = blockIdx.x;
    const float*  gi  = dir ? gi_b  : gi_f;
    const __bf16* Whh = dir ? Whh_b : Whh_f;
    const float*  bhh = dir ? bhh_b : bhh_f;

    const int tid    = threadIdx.x;
    const int wave   = tid >> 5;
    const int lane   = tid & 31;
    const int lane15 = lane & 15;
    const int hi     = (lane >= 16) ? 1 : 0;

    const __bf16* Br[3];
#pragma unroll
    for (int nt = 0; nt < 3; ++nt)
        Br[nt] = Whh + (size_t)((wave * 3 + nt) * 16 + lane15) * HH;
    const float* Ar[2] = { h_s + (size_t)lane15 * HH, h_s + (size_t)(16 + lane15) * HH };

    for (int i = tid; i < BB * HH; i += 1024) h_s[i] = 0.0f;
    __syncthreads();

    for (int t = 0; t < SS; ++t) {
        // ---- gh = h @ Whh^T + bhh ----
        f32x8 acc[2][3];
        const f32x8 zero = {};
#pragma unroll
        for (int mt = 0; mt < 2; ++mt)
#pragma unroll
            for (int nt = 0; nt < 3; ++nt) acc[mt][nt] = zero;

        for (int k0 = 0; k0 < HH; k0 += 32) {
            bf16x16 a[2];
#pragma unroll
            for (int mt = 0; mt < 2; ++mt) load_a_frag(a[mt], Ar[mt], k0, hi); // ds_load
            bf16x16 b[3];
#pragma unroll
            for (int nt = 0; nt < 3; ++nt) load_b_frag(b[nt], Br[nt], k0, hi);
#pragma unroll
            for (int mt = 0; mt < 2; ++mt)
#pragma unroll
                for (int nt = 0; nt < 3; ++nt)
                    acc[mt][nt] = __builtin_amdgcn_wmma_f32_16x16x32_bf16(
                        false, a[mt], false, b[nt], (short)0, acc[mt][nt], false, false);
        }
#pragma unroll
        for (int nt = 0; nt < 3; ++nt) {
            const int col = (wave * 3 + nt) * 16 + lane15;
            const float bv = bhh[col];
#pragma unroll
            for (int mt = 0; mt < 2; ++mt)
#pragma unroll
                for (int v = 0; v < 8; ++v) {
                    int row = mt * 16 + v + (hi ? 8 : 0);
                    gh_s[row * (3 * HH) + col] = acc[mt][nt][v] + bv;
                }
        }
        __syncthreads();

        // ---- GRU gates, update h, write encoder_hidden ----
        const float* gi_t = gi + (size_t)t * BB * (3 * HH);
        for (int i = tid; i < BB * HH; i += 1024) {
            const int b = i >> 9;
            const int j = i & (HH - 1);
            const float* gr = gi_t + (size_t)b * (3 * HH);
            const float i_r = gr[j];
            const float i_z = gr[HH + j];
            const float i_n = gr[2 * HH + j];
            const float h_r = gh_s[b * (3 * HH) + j];
            const float h_z = gh_s[b * (3 * HH) + HH + j];
            const float h_n = gh_s[b * (3 * HH) + 2 * HH + j];
            const float r = sigm(i_r + h_r);
            const float z = sigm(i_z + h_z);
            const float n = tanhf(i_n + r * h_n);
            const float hp = h_s[i];
            const float hn = (1.0f - z) * n + z * hp;
            h_s[i] = hn;
            const int pos = dir ? (SS - 1 - t) : t;
            enc_hid[((size_t)b * SS + pos) * (2 * HH) + dir * HH + j] = hn;
            if (t == SS - 1) enc_final[b * (2 * HH) + dir * HH + j] = hn;
        }
        __syncthreads();
    }
}

// ---------------------------------------------------------------------
// Small helper kernels
// ---------------------------------------------------------------------
__global__ void k_f2bf(__bf16* __restrict__ dst, const float* __restrict__ src, int n)
{
    int i = blockIdx.x * blockDim.x + threadIdx.x;
    if (i < n) dst[i] = f2bf(src[i]);
}

__global__ void k_embed_x(float* __restrict__ x, const float* __restrict__ embed_W,
                          const int* __restrict__ src)
{
    int i = blockIdx.x * blockDim.x + threadIdx.x;   // (s,b,e)
    if (i >= SS * BB * EE) return;
    int e = i % EE;
    int sb = i / EE;
    int b = sb % BB;
    int s = sb / BB;
    x[i] = embed_W[(size_t)src[b * SS + s] * EE + e];
}

__global__ void k_gather_up(float* __restrict__ u_buf, float* __restrict__ p_buf,
                            const float* __restrict__ user_W, const float* __restrict__ prod_W,
                            const int* __restrict__ user, const int* __restrict__ product)
{
    int i = blockIdx.x * blockDim.x + threadIdx.x;
    if (i >= BB * AA) return;
    int b = i >> 6, j = i & 63;
    u_buf[i] = user_W[(size_t)user[b] * AA + j];
    p_buf[i] = prod_W[(size_t)product[b] * AA + j];
}

__global__ void k_copy(float* __restrict__ dst, const float* __restrict__ src, int n)
{
    int i = blockIdx.x * blockDim.x + threadIdx.x;
    if (i < n) dst[i] = src[i];
}

__global__ void k_build_rnn_in(float* __restrict__ rnn_in, const float* __restrict__ embed_W,
                               const int* __restrict__ trg, const float* __restrict__ ch,
                               const float* __restrict__ u, const float* __restrict__ p, int t)
{
    int i = blockIdx.x * blockDim.x + threadIdx.x;
    if (i >= BB * 896) return;
    int b = i / 896, j = i % 896;
    float v;
    if (j < EE) {
        int tok = (t == 0) ? 1 : trg[b * LL + (t - 1)];
        v = embed_W[(size_t)tok * EE + j];
    } else if (j < EE + HH) v = ch[b * HH + (j - EE)];
    else if (j < EE + HH + AA) v = u[b * AA + (j - EE - HH)];
    else v = p[b * AA + (j - EE - HH - AA)];
    rnn_in[i] = v;
}

__global__ void k_gru_gate(const float* __restrict__ gi, const float* __restrict__ gh,
                           float* __restrict__ h)
{
    int i = blockIdx.x * blockDim.x + threadIdx.x;
    if (i >= BB * HH) return;
    int b = i >> 9, j = i & (HH - 1);
    const float i_r = gi[b * (3 * HH) + j];
    const float i_z = gi[b * (3 * HH) + HH + j];
    const float i_n = gi[b * (3 * HH) + 2 * HH + j];
    const float h_r = gh[b * (3 * HH) + j];
    const float h_z = gh[b * (3 * HH) + HH + j];
    const float h_n = gh[b * (3 * HH) + 2 * HH + j];
    const float r = sigm(i_r + h_r);
    const float z = sigm(i_z + h_z);
    const float n = tanhf(i_n + r * h_n);
    h[i] = (1.0f - z) * n + z * h[i];
}

__global__ void k_attention(const float* __restrict__ qW, const float* __restrict__ proj_key,
                            const float* __restrict__ enc_hid, const float* __restrict__ energy,
                            const int* __restrict__ mask,
                            float* __restrict__ alphas, float* __restrict__ context)
{
    __shared__ float q_s[HH];
    __shared__ float sc_s[SS];
    __shared__ float red[512];
    const int b = blockIdx.x, tid = threadIdx.x;  // 512 threads
    q_s[tid] = qW[b * HH + tid];
    __syncthreads();
    for (int s = tid; s < SS; s += 512) {
        const float* pk = proj_key + ((size_t)b * SS + s) * HH;
        float acc = 0.0f;
        for (int h = 0; h < HH; ++h) acc += energy[h] * tanhf(q_s[h] + pk[h]);
        sc_s[s] = mask[b * SS + s] ? acc : -INFINITY;
    }
    __syncthreads();
    float m = -INFINITY;
    for (int s = tid; s < SS; s += 512) m = fmaxf(m, sc_s[s]);
    red[tid] = m; __syncthreads();
    for (int o = 256; o > 0; o >>= 1) { if (tid < o) red[tid] = fmaxf(red[tid], red[tid + o]); __syncthreads(); }
    m = red[0]; __syncthreads();
    float ssum = 0.0f;
    for (int s = tid; s < SS; s += 512) { float e = __expf(sc_s[s] - m); sc_s[s] = e; ssum += e; }
    red[tid] = ssum; __syncthreads();
    for (int o = 256; o > 0; o >>= 1) { if (tid < o) red[tid] += red[tid + o]; __syncthreads(); }
    const float inv = 1.0f / red[0]; __syncthreads();
    for (int s = tid; s < SS; s += 512) { float a = sc_s[s] * inv; sc_s[s] = a; alphas[b * SS + s] = a; }
    __syncthreads();
    for (int d = tid; d < 2 * HH; d += 512) {
        float acc = 0.0f;
        for (int s = 0; s < SS; ++s) acc += sc_s[s] * enc_hid[((size_t)b * SS + s) * (2 * HH) + d];
        context[b * (2 * HH) + d] = acc;
    }
}

__global__ void k_concat_hc(float* __restrict__ hc, const float* __restrict__ h,
                            const float* __restrict__ ctx)
{
    int i = blockIdx.x * blockDim.x + threadIdx.x;
    if (i >= BB * (3 * HH)) return;
    int b = i / (3 * HH), j = i % (3 * HH);
    hc[i] = (j < HH) ? h[b * HH + j] : ctx[b * (2 * HH) + (j - HH)];
}

__global__ void k_genp(const float* __restrict__ context, const float* __restrict__ h,
                       const float* __restrict__ rnn_in, const float* __restrict__ genp_W,
                       const float* __restrict__ genp_b, float* __restrict__ gp)
{
    __shared__ float red[256];
    const int b = blockIdx.x, tid = threadIdx.x;   // 256 threads
    float acc = 0.0f;
    for (int i = tid; i < 3 * HH + EE; i += 256) {
        float x;
        if (i < 2 * HH) x = context[b * (2 * HH) + i];
        else if (i < 3 * HH) x = h[b * HH + (i - 2 * HH)];
        else x = rnn_in[b * 896 + (i - 3 * HH)];   // pe lives in rnn_in[:, :E]
        acc += genp_W[i] * x;
    }
    red[tid] = acc; __syncthreads();
    for (int o = 128; o > 0; o >>= 1) { if (tid < o) red[tid] += red[tid + o]; __syncthreads(); }
    if (tid == 0) gp[b] = sigm(red[0] + genp_b[0]);
}

__global__ void k_softmax_mix(const float* __restrict__ logits, const float* __restrict__ gp,
                              float* __restrict__ out, int t)
{
    __shared__ float red[1024];
    const int b = blockIdx.x, tid = threadIdx.x;   // 1024 threads
    const float* lr = logits + (size_t)b * VV;
    float* orow = out + ((size_t)b * LL + t) * VV;
    float m = -INFINITY;
    for (int v = tid; v < VV; v += 1024) m = fmaxf(m, lr[v]);
    red[tid] = m; __syncthreads();
    for (int o = 512; o > 0; o >>= 1) { if (tid < o) red[tid] = fmaxf(red[tid], red[tid + o]); __syncthreads(); }
    m = red[0]; __syncthreads();
    float ssum = 0.0f;
    for (int v = tid; v < VV; v += 1024) ssum += __expf(lr[v] - m);
    red[tid] = ssum; __syncthreads();
    for (int o = 512; o > 0; o >>= 1) { if (tid < o) red[tid] += red[tid + o]; __syncthreads(); }
    const float scale = gp[b] / red[0];
    for (int v = tid; v < VV; v += 1024) orow[v] = __expf(lr[v] - m) * scale;
}

__global__ void k_scatter_copy(float* __restrict__ out, const int* __restrict__ src,
                               const float* __restrict__ alphas, const float* __restrict__ gp,
                               int t)
{
    int i = blockIdx.x * blockDim.x + threadIdx.x;
    if (i >= BB * SS) return;
    int b = i / SS;
    float w = (1.0f - gp[b]) * alphas[i];
    atomicAdd(out + ((size_t)b * LL + t) * VV + src[i], w);
}

// ---------------------------------------------------------------------
// Host side
// ---------------------------------------------------------------------
static inline int cdiv(long long a, long long b) { return (int)((a + b - 1) / b); }

static void gemm(const float* A, int lda, const void* Bw, const float* bias,
                 float* C, int ldc, int M, int N, int K, int act, int revS, int revB,
                 hipStream_t s)
{
    if (M >= 128) {
        // 2x4 register blocking: high A/B reuse for the large GEMMs
        int waves = (M / 32) * cdiv(N, 64);
        k_wmma_gemm<2, 4><<<cdiv(waves, 8), 256, 0, s>>>(
            A, lda, (const __bf16*)Bw, bias, C, ldc, M, N, K, act, revS, revB);
    } else {
        // 2x2 blocking: keeps wave-level parallelism up for M=32 GEMMs
        int waves = (M / 32) * cdiv(N, 32);
        k_wmma_gemm<2, 2><<<cdiv(waves, 8), 256, 0, s>>>(
            A, lda, (const __bf16*)Bw, bias, C, ldc, M, N, K, act, revS, revB);
    }
}

static void cvt(void* dst, const float* src, long long n, hipStream_t s)
{
    k_f2bf<<<cdiv(n, 256), 256, 0, s>>>((__bf16*)dst, src, (int)n);
}

extern "C" void kernel_launch(void* const* d_in, const int* in_sizes, int n_in,
                              void* d_out, int out_size, void* d_ws, size_t ws_size,
                              hipStream_t stream)
{
    const int*   src      = (const int*)d_in[0];
    const int*   trg      = (const int*)d_in[1];
    const int*   user     = (const int*)d_in[2];
    const int*   product  = (const int*)d_in[3];
    const int*   src_mask = (const int*)d_in[4];
    const float* embed_W  = (const float*)d_in[6];
    const float* user_W   = (const float*)d_in[7];
    const float* prod_W   = (const float*)d_in[8];
    const float* Wih_f    = (const float*)d_in[9];
    const float* Whh_f    = (const float*)d_in[10];
    const float* bih_f    = (const float*)d_in[11];
    const float* bhh_f    = (const float*)d_in[12];
    const float* Wih_b    = (const float*)d_in[13];
    const float* Whh_b    = (const float*)d_in[14];
    const float* bih_b    = (const float*)d_in[15];
    const float* bhh_b    = (const float*)d_in[16];
    const float* init_W   = (const float*)d_in[17];
    const float* init_b   = (const float*)d_in[18];
    const float* key_W    = (const float*)d_in[19];
    const float* query_W  = (const float*)d_in[20];
    const float* energy_W = (const float*)d_in[21];
    const float* dec_Wih  = (const float*)d_in[22];
    const float* dec_Whh  = (const float*)d_in[23];
    const float* dec_bih  = (const float*)d_in[24];
    const float* dec_bhh  = (const float*)d_in[25];
    const float* ch_W     = (const float*)d_in[26];
    const float* genp_W   = (const float*)d_in[27];
    const float* genp_b   = (const float*)d_in[28];
    const float* gen_W    = (const float*)d_in[29];
    float* out = (float*)d_out;

    // bump allocator over workspace
    char* w = (char*)d_ws;
    auto alloc = [&](size_t bytes) -> void* {
        void* p = (void*)w;
        w += (bytes + 255) & ~(size_t)255;
        return p;
    };
    void* wbWihF  = alloc((size_t)3 * HH * EE * 2);
    void* wbWihB  = alloc((size_t)3 * HH * EE * 2);
    void* wbWhhF  = alloc((size_t)3 * HH * HH * 2);
    void* wbWhhB  = alloc((size_t)3 * HH * HH * 2);
    void* wbInit  = alloc((size_t)HH * 2 * HH * 2);
    void* wbKey   = alloc((size_t)HH * 2 * HH * 2);
    void* wbQuery = alloc((size_t)HH * HH * 2);
    void* wbDecWih= alloc((size_t)3 * HH * 896 * 2);
    void* wbDecWhh= alloc((size_t)3 * HH * HH * 2);
    void* wbCh    = alloc((size_t)HH * 3 * HH * 2);
    void* wbGen   = alloc((size_t)VV * HH * 2);
    float* x       = (float*)alloc((size_t)SS * BB * EE * 4);
    float* giF     = (float*)alloc((size_t)SS * BB * 3 * HH * 4);
    float* giB     = (float*)alloc((size_t)SS * BB * 3 * HH * 4);
    float* encHid  = (float*)alloc((size_t)BB * SS * 2 * HH * 4);
    float* encFin  = (float*)alloc((size_t)BB * 2 * HH * 4);
    float* projKey = (float*)alloc((size_t)BB * SS * HH * 4);
    float* hBuf    = (float*)alloc((size_t)BB * HH * 4);
    float* chBuf   = (float*)alloc((size_t)BB * HH * 4);
    float* uBuf    = (float*)alloc((size_t)BB * AA * 4);
    float* pBuf    = (float*)alloc((size_t)BB * AA * 4);
    float* rnnIn   = (float*)alloc((size_t)BB * 896 * 4);
    float* dgi     = (float*)alloc((size_t)BB * 3 * HH * 4);
    float* dgh     = (float*)alloc((size_t)BB * 3 * HH * 4);
    float* qWbuf   = (float*)alloc((size_t)BB * HH * 4);
    float* alphas  = (float*)alloc((size_t)BB * SS * 4);
    float* ctx     = (float*)alloc((size_t)BB * 2 * HH * 4);
    float* hc      = (float*)alloc((size_t)BB * 3 * HH * 4);
    float* gp      = (float*)alloc((size_t)BB * 4);
    float* logits  = (float*)alloc((size_t)BB * VV * 4);

    // 1. weight conversion to bf16
    cvt(wbWihF,  Wih_f,   (long long)3 * HH * EE, stream);
    cvt(wbWihB,  Wih_b,   (long long)3 * HH * EE, stream);
    cvt(wbWhhF,  Whh_f,   (long long)3 * HH * HH, stream);
    cvt(wbWhhB,  Whh_b,   (long long)3 * HH * HH, stream);
    cvt(wbInit,  init_W,  (long long)HH * 2 * HH, stream);
    cvt(wbKey,   key_W,   (long long)HH * 2 * HH, stream);
    cvt(wbQuery, query_W, (long long)HH * HH, stream);
    cvt(wbDecWih,dec_Wih, (long long)3 * HH * 896, stream);
    cvt(wbDecWhh,dec_Whh, (long long)3 * HH * HH, stream);
    cvt(wbCh,    ch_W,    (long long)HH * 3 * HH, stream);
    cvt(wbGen,   gen_W,   (long long)VV * HH, stream);

    // 2. embeddings, input GEMMs (M = S*B = 12800)
    k_embed_x<<<cdiv((long long)SS * BB * EE, 256), 256, 0, stream>>>(x, embed_W, src);
    gemm(x, EE, wbWihF, bih_f, giF, 3 * HH, SS * BB, 3 * HH, EE, 0, 0, 0, stream);
    gemm(x, EE, wbWihB, bih_b, giB, 3 * HH, SS * BB, 3 * HH, EE, 0, SS, BB, stream);

    // 3. fused bidirectional GRU scan (2 persistent blocks, 256 KB LDS each)
    k_enc_scan<<<2, 1024, (size_t)(BB * HH + BB * 3 * HH) * 4, stream>>>(
        giF, giB, (const __bf16*)wbWhhF, (const __bf16*)wbWhhB, bhh_f, bhh_b,
        encHid, encFin);

    // 4. init hidden, proj_key
    gemm(encFin, 2 * HH, wbInit, init_b, hBuf, HH, BB, HH, 2 * HH, 1, 0, 0, stream);
    k_copy<<<cdiv(BB * HH, 256), 256, 0, stream>>>(chBuf, hBuf, BB * HH);
    gemm(encHid, 2 * HH, wbKey, nullptr, projKey, HH, BB * SS, HH, 2 * HH, 0, 0, 0, stream);
    k_gather_up<<<cdiv(BB * AA, 256), 256, 0, stream>>>(uBuf, pBuf, user_W, prod_W, user, product);

    // 5. decoder loop
    for (int t = 0; t < LL; ++t) {
        k_build_rnn_in<<<cdiv(BB * 896, 256), 256, 0, stream>>>(rnnIn, embed_W, trg,
                                                                chBuf, uBuf, pBuf, t);
        gemm(rnnIn, 896, wbDecWih, dec_bih, dgi, 3 * HH, BB, 3 * HH, 896, 0, 0, 0, stream);
        gemm(hBuf, HH, wbDecWhh, dec_bhh, dgh, 3 * HH, BB, 3 * HH, HH, 0, 0, 0, stream);
        k_gru_gate<<<cdiv(BB * HH, 256), 256, 0, stream>>>(dgi, dgh, hBuf);
        gemm(hBuf, HH, wbQuery, nullptr, qWbuf, HH, BB, HH, HH, 0, 0, 0, stream);
        k_attention<<<BB, 512, 0, stream>>>(qWbuf, projKey, encHid, energy_W, src_mask,
                                            alphas, ctx);
        k_concat_hc<<<cdiv(BB * 3 * HH, 256), 256, 0, stream>>>(hc, hBuf, ctx);
        gemm(hc, 3 * HH, wbCh, nullptr, chBuf, HH, BB, HH, 3 * HH, 1, 0, 0, stream);
        gemm(chBuf, HH, wbGen, nullptr, logits, VV, BB, VV, HH, 0, 0, 0, stream);
        k_genp<<<BB, 256, 0, stream>>>(ctx, hBuf, rnnIn, genp_W, genp_b, gp);
        k_softmax_mix<<<BB, 1024, 0, stream>>>(logits, gp, out, t);
        k_scatter_copy<<<cdiv(BB * SS, 256), 256, 0, stream>>>(out, src, alphas, gp, t);
    }
}
